// RNNModel_45887430590809
// MI455X (gfx1250) — matrix operand order repaired and checked by reading, not compiled, not run on hardware
//
#include <hip/hip_runtime.h>
#include <math.h>
#include <stdint.h>

typedef __attribute__((ext_vector_type(16))) __bf16 v16bf;
typedef __attribute__((ext_vector_type(8)))  __bf16 v8bf;
typedef __attribute__((ext_vector_type(8)))  float  v8f;

// global (addrspace 1) pointer types so laundered loads stay global_load_*
typedef const __attribute__((address_space(1))) __bf16 *gbf_ptr;
typedef const __attribute__((address_space(1))) v16bf  *gv16_ptr;

#define VOCAB_  50000
#define B_      1024
#define T_      120
#define E_      50
#define H_      300
#define C_      5
#define KPAD    352            // ceil((50+300)/32)*32 = 11*32
#define NPAD    1280           // 4 gates * 320 (300 padded to 320 = 20 tiles)
#define KT      11             // K tiles of 32
#define NT      80             // N tiles of 16
#define BM      16             // batch rows per workgroup
#define NWAVES  10
#define NTHREADS (NWAVES*32)   // 320
#define NBLK    (B_/BM)        // 64

#define WS_WFRAG_OFF 1024      // bytes; partials live at offset 0 (64*2 floats)

__device__ __forceinline__ float sigf(float x) { return 1.0f / (1.0f + __expf(-x)); }

__device__ __forceinline__ float fast_tanh(float x) {
#if __has_builtin(__builtin_amdgcn_tanhf)
  return __builtin_amdgcn_tanhf(x);
#elif __has_builtin(__builtin_amdgcn_tanh_f32)
  return __builtin_amdgcn_tanh_f32(x);
#else
  float r;
  // native gfx1250 TRANS op; v_nop satisfies the 1-instruction TRANS hazard
  asm("v_tanh_f32 %0, %1\n\tv_nop" : "=v"(r) : "v"(x));
  return r;
#endif
}

// ---------------------------------------------------------------------------
// Pre-swizzle [Wx;Wh] (fp32) into WMMA-B fragment-ready bf16 layout:
// Wfrag[((kt*NT + nt)*32 + lane)*16 + e]  holds  W[k][n] with
//   n = nt*16 + (lane&15),  k = kt*32 + (lane>=16 ? 16 : 0) + e
// ---------------------------------------------------------------------------
__global__ void prep_weights(const float* __restrict__ Wx,
                             const float* __restrict__ Wh,
                             __bf16* __restrict__ Wfrag) {
  int idx = blockIdx.x * blockDim.x + threadIdx.x;
  const int total = KT * NT * 32 * 16;
  if (idx >= total) return;
  int e    = idx & 15;
  int lane = (idx >> 4) & 31;
  int tile = idx >> 9;
  int kt = tile / NT, nt = tile % NT;
  int k = kt * 32 + ((lane & 16) ? 16 : 0) + e;
  int n = nt * 16 + (lane & 15);
  int gate = n / 320, unit = n % 320;
  float v = 0.0f;
  if (unit < H_ && k < E_ + H_) {
    int col = gate * H_ + unit;                 // gate order i,j,f,o in 4H
    v = (k < E_) ? Wx[k * (4 * H_) + col] : Wh[(k - E_) * (4 * H_) + col];
  }
  Wfrag[idx] = (__bf16)v;
}

// ---------------------------------------------------------------------------
// Persistent recurrent kernel: one workgroup = 16 batch rows for all 120 steps.
// A operand in LDS = [x_t (K 0..49) ; h (K 50..349) ; zero pad], bf16.
// Gates z computed with v_wmma_f32_16x16x32_bf16; c-state register resident.
// ---------------------------------------------------------------------------
__global__ __launch_bounds__(NTHREADS)
void lstm_loss_kernel(const int* __restrict__ tokens,
                      const int* __restrict__ labels,
                      const unsigned char* __restrict__ mask,
                      const float* __restrict__ emb,
                      const float* __restrict__ b,
                      const float* __restrict__ U,
                      const float* __restrict__ b2,
                      const __bf16* __restrict__ Wfrag,
                      float* __restrict__ partial) {
  __shared__ __align__(16) __bf16 A_lds[BM * KPAD];    // 11.0 KB
  __shared__ float h_lds[BM * H_];                      // 18.75 KB
  __shared__ float U_lds[H_ * C_ + C_];                 // 5.9 KB (U + b2)
  __shared__ float logit_lds[BM * C_];
  __shared__ float red[BM * 2];

  const int tid  = threadIdx.x;
  const int wave = tid >> 5;
  const int lane = tid & 31;
  const int hi   = (lane >> 4) & 1;   // K-half / M-half selector
  const int am   = lane & 15;         // A row (M) for this lane
  const int brow0 = blockIdx.x * BM;

  // zero all of A (h region stays zero for t=0; x rewritten each step)
  for (int i = tid; i < BM * KPAD; i += NTHREADS) A_lds[i] = (__bf16)0.0f;
  // preload U and b2 into LDS (reused 120x)
  for (int i = tid; i < H_ * C_; i += NTHREADS) U_lds[i] = U[i];
  if (tid < C_) U_lds[H_ * C_ + tid] = b2[tid];

  // per-lane gate bias: wave owns N-tiles nt = g*20 + (2*wave + p)
  float bias_g[4][2];
  #pragma unroll
  for (int g = 0; g < 4; ++g)
    #pragma unroll
    for (int p = 0; p < 2; ++p) {
      int u = (wave * 2 + p) * 16 + am;
      bias_g[g][p] = (u < H_) ? b[g * H_ + u] : 0.0f;
    }

  float c_reg[2][8];
  #pragma unroll
  for (int p = 0; p < 2; ++p)
    #pragma unroll
    for (int r = 0; r < 8; ++r) c_reg[p][r] = 0.0f;

  float ce_acc = 0.0f, cnt_acc = 0.0f;   // used by tid < 16
  const v8f vzero = {};

  __syncthreads();

  for (int t = 0; t < T_; ++t) {
    // ---- phase 0: embedding gather for x_t -> A_lds[:, 0..49] ----
    for (int i = tid; i < BM * E_; i += NTHREADS) {
      int m = i / E_, k = i % E_;
      int tok = tokens[(brow0 + m) * T_ + t];
      A_lds[m * KPAD + k] = (__bf16)emb[tok * E_ + k];
    }
    __syncthreads();

    // Launder the weight base pointer so LICM cannot hoist the (L2-resident)
    // fragment loads out of the time loop and spill 88 fragments to scratch.
    // Cast back into addrspace(1) so loads stay global_load_b128 (pure LOADcnt).
    uintptr_t wq = (uintptr_t)Wfrag;
    asm volatile("" : "+s"(wq));
    gbf_ptr wbase = (gbf_ptr)wq;

    // ---- phase 1: z = [x_t ; h] @ [Wx;Wh]  via WMMA ----
    v8f acc[4][2];
    #pragma unroll
    for (int g = 0; g < 4; ++g)
      #pragma unroll
      for (int p = 0; p < 2; ++p) acc[g][p] = vzero;

    for (int kt = 0; kt < KT; ++kt) {
      // A fragment (contiguous K order, matches Wfrag): lane<16 -> K {0..15}, else {16..31}
      const __bf16* ap = &A_lds[am * KPAD + kt * 32 + hi * 16];
      v8bf a_lo = *(const v8bf*)ap;
      v8bf a_hi = *(const v8bf*)(ap + 8);
      v16bf afrag = __builtin_shufflevector(a_lo, a_hi,
          0, 1, 2, 3, 4, 5, 6, 7, 8, 9, 10, 11, 12, 13, 14, 15);

      // batch all 8 B-fragment loads first (one clause, staged waits) ...
      v16bf bfrag[4][2];
      #pragma unroll
      for (int g = 0; g < 4; ++g)
        #pragma unroll
        for (int p = 0; p < 2; ++p) {
          int nt = g * 20 + (wave * 2 + p);
          bfrag[g][p] = *(gv16_ptr)(wbase + (((kt * NT + nt) * 32 + lane) << 4));
        }
      // ... then issue the 8 WMMAs
      #pragma unroll
      for (int g = 0; g < 4; ++g)
        #pragma unroll
        for (int p = 0; p < 2; ++p)
          acc[g][p] = __builtin_amdgcn_wmma_f32_16x16x32_bf16(
              false, afrag, false, bfrag[g][p], (short)0, acc[g][p], false, false);
    }
    __syncthreads();   // all WMMA reads of A done before h overwrite

    // ---- phase 2: LSTM nonlinearity, fully register resident ----
    // C/D layout: VGPR r -> m = r + 8*hi, column = lane&15
    #pragma unroll
    for (int p = 0; p < 2; ++p) {
      int u = (wave * 2 + p) * 16 + am;
      bool valid = (u < H_);
      #pragma unroll
      for (int r = 0; r < 8; ++r) {
        int m = r + hi * 8;
        float zi = acc[0][p][r] + bias_g[0][p];
        float zj = acc[1][p][r] + bias_g[1][p];
        float zf = acc[2][p][r] + bias_g[2][p];
        float zo = acc[3][p][r] + bias_g[3][p];
        float c = sigf(zf + 1.0f) * c_reg[p][r] + sigf(zi) * fast_tanh(zj);
        float h = sigf(zo) * fast_tanh(c);
        c_reg[p][r] = c;
        if (valid) {
          h_lds[m * H_ + u] = h;
          A_lds[m * KPAD + E_ + u] = (__bf16)h;   // feed next step's recurrence
        }
      }
    }
    __syncthreads();

    // ---- phase 3: logits = h @ U + b2 (C=5), then CE ----
    if (tid < BM * C_) {
      int m = tid / C_, cls = tid % C_;
      float dot = U_lds[H_ * C_ + cls];
      #pragma unroll 4
      for (int u = 0; u < H_; ++u) dot += h_lds[m * H_ + u] * U_lds[u * C_ + cls];
      logit_lds[tid] = dot;
    }
    __syncthreads();

    if (tid < BM) {
      int m = tid;
      float lg[C_], mx = -1e30f;
      #pragma unroll
      for (int cc = 0; cc < C_; ++cc) { lg[cc] = logit_lds[m * C_ + cc]; mx = fmaxf(mx, lg[cc]); }
      float s = 0.0f;
      #pragma unroll
      for (int cc = 0; cc < C_; ++cc) s += __expf(lg[cc] - mx);
      float lse = mx + __logf(s);
      int gi = (brow0 + m) * T_ + t;
      if (mask[gi]) { ce_acc += lse - lg[labels[gi]]; cnt_acc += 1.0f; }
    }
    // next iteration's first barrier orders CE reads vs. logit_lds overwrite
  }

  // deterministic per-workgroup reduction
  if (tid < BM) { red[tid * 2] = ce_acc; red[tid * 2 + 1] = cnt_acc; }
  __syncthreads();
  if (tid == 0) {
    float s = 0.0f, c = 0.0f;
    for (int m = 0; m < BM; ++m) { s += red[m * 2]; c += red[m * 2 + 1]; }
    partial[blockIdx.x * 2]     = s;
    partial[blockIdx.x * 2 + 1] = c;
  }
}

__global__ void finalize_loss(const float* __restrict__ partial,
                              float* __restrict__ out) {
  float s = 0.0f, c = 0.0f;
  for (int i = 0; i < NBLK; ++i) { s += partial[i * 2]; c += partial[i * 2 + 1]; }
  out[0] = s / c;
}

extern "C" void kernel_launch(void* const* d_in, const int* in_sizes, int n_in,
                              void* d_out, int out_size, void* d_ws, size_t ws_size,
                              hipStream_t stream) {
  const int*           tokens = (const int*)d_in[0];
  const int*           labels = (const int*)d_in[1];
  const unsigned char* mask   = (const unsigned char*)d_in[2];  // jnp bool_ = 1 byte
  const float*         emb    = (const float*)d_in[3];
  const float*         Wx     = (const float*)d_in[4];
  const float*         Wh     = (const float*)d_in[5];
  const float*         b      = (const float*)d_in[6];
  const float*         U      = (const float*)d_in[7];
  const float*         b2     = (const float*)d_in[8];

  float*  partial = (float*)d_ws;                               // 64*2 floats
  __bf16* Wfrag   = (__bf16*)((char*)d_ws + WS_WFRAG_OFF);      // 11*80*512 bf16 = 880 KB

  const int total = KT * NT * 32 * 16;
  prep_weights<<<(total + 255) / 256, 256, 0, stream>>>(Wx, Wh, Wfrag);
  lstm_loss_kernel<<<NBLK, NTHREADS, 0, stream>>>(tokens, labels, mask, emb,
                                                  b, U, b2, Wfrag, partial);
  finalize_loss<<<1, 1, 0, stream>>>(partial, (float*)d_out);
}
